// SelfAttention_35115652612730
// MI455X (gfx1250) — matrix operand order repaired and checked by reading, not compiled, not run on hardware
//
#include <hip/hip_runtime.h>

typedef __attribute__((ext_vector_type(16))) _Float16 v16h;
typedef __attribute__((ext_vector_type(8)))  float    v8f;

#define NPIX  9216
#define CDIM  256
#define CQN   32
#define LOG2E 1.4426950408889634f

union Frag16 {
  v16h v;
  _Float16 h[16];
  unsigned u[8];
};

// ---------------------------------------------------------------------------
// Kernel 1: QKV projection  qkv[o,n] = sum_c w[o,c] * x[b,c,n]
// One wave per (o-tile 16, n-tile 16, b). 2*20*576 = 23040 wave-tiles.
// q written pre-scaled by log2(e) so attention softmax can run in exp2 domain.
// q,k layout: [b][n][32] f16 (64B contiguous rows); v layout: [b][c][n] f16.
// ---------------------------------------------------------------------------
__global__ __launch_bounds__(256)
void qkv_proj_kernel(const float* __restrict__ x, const float* __restrict__ w,
                     _Float16* __restrict__ qh, _Float16* __restrict__ kh,
                     _Float16* __restrict__ vh)
{
  const int lane  = threadIdx.x & 31;
  const int wv    = threadIdx.x >> 5;
  const int tile  = blockIdx.x * 8 + wv;
  const int nt    = tile % 576;
  const int ot    = (tile / 576) % 20;
  const int b     = tile / (576 * 20);
  const int lhalf = lane & 15;
  const bool lo   = lane < 16;

  const int orow  = ot * 16 + lhalf;                 // A-matrix M row (this lane)
  const int ncol  = nt * 16 + lhalf;                 // B/D-matrix N column (this lane)
  const float* wrow = w + (size_t)orow * CDIM;
  const float* xb   = x + (size_t)b * CDIM * NPIX;

  v8f acc = {};
  for (int c0 = 0; c0 < CDIM; c0 += 32) {
    Frag16 A, Bf;
#pragma unroll
    for (int n = 0; n < 8; ++n) {
      // A (w) 16x32 f16 layout: lanes 0-15 K in {0-7,16-23}, lanes 16-31 {8-15,24-31}
      int ka = lo ? (n < 4 ? 2*n : 2*n + 8) : (n < 4 ? 2*n + 8 : 2*n + 16);
      float2 wp = *(const float2*)(wrow + c0 + ka);
      A.h[2*n]   = (_Float16)wp.x;
      A.h[2*n+1] = (_Float16)wp.y;
      // B (x) 32x16 f16 layout: lanes 0-15 K in {0-7,16-23}, lanes 16-31 {8-15,24-31}
      int kb = lo ? (n < 4 ? 2*n : 2*n + 8) : (n < 4 ? 2*n + 8 : 2*n + 16);
      Bf.h[2*n]   = (_Float16)xb[(size_t)(c0 + kb)     * NPIX + ncol];
      Bf.h[2*n+1] = (_Float16)xb[(size_t)(c0 + kb + 1) * NPIX + ncol];
    }
    acc = __builtin_amdgcn_wmma_f32_16x16x32_f16(false, A.v, false, Bf.v,
                                                 (short)0, acc, false, false);
  }

#pragma unroll
  for (int r = 0; r < 8; ++r) {
    int o2 = ot * 16 + r + (lo ? 0 : 8);             // D layout: lanes>=16 hold M=r+8
    float val = acc[r];
    if (o2 < CQN) {
      qh[((size_t)b * NPIX + ncol) * CQN + o2] = (_Float16)(val * LOG2E);
    } else if (o2 < 2 * CQN) {
      kh[((size_t)b * NPIX + ncol) * CQN + (o2 - CQN)] = (_Float16)val;
    } else {
      vh[((size_t)b * CDIM + (o2 - 2 * CQN)) * NPIX + ncol] = (_Float16)val;
    }
  }
}

// ---------------------------------------------------------------------------
// Async global->LDS tile fetch (gfx1250 ASYNCcnt path, GV addressing mode).
// Each active lane copies 16B directly into LDS, no VGPR staging.
// q tile: 2KB contiguous (waves 0-3 only -> wave-uniform predicate, no EXEC
// divergence within a wave). v tile: 256 rows x 64B.
// ---------------------------------------------------------------------------
__device__ __forceinline__ void issue_tile_loads(const uint4* qg4, const uint4* vg4,
                                                 int i0, unsigned* sQ, unsigned* sV,
                                                 int t)
{
  if (t < 128) {
    unsigned lq = (unsigned)(uintptr_t)&sQ[t * 4];
    unsigned long long ga = (unsigned long long)(uintptr_t)&qg4[i0 * 4 + t];
    asm volatile("global_load_async_to_lds_b128 %0, %1, off"
                 :: "v"(lq), "v"(ga) : "memory");
  }
#pragma unroll
  for (int kk = 0; kk < 4; ++kk) {
    int ch = t + kk * 256;                        // 0..1023 uint4 chunks
    int c = ch >> 2, part = ch & 3;
    unsigned lv = (unsigned)(uintptr_t)&sV[ch * 4];
    unsigned long long ga =
        (unsigned long long)(uintptr_t)&vg4[(size_t)c * (NPIX / 8) + (i0 >> 3) + part];
    asm volatile("global_load_async_to_lds_b128 %0, %1, off"
                 :: "v"(lv), "v"(ga) : "memory");
  }
}

// ---------------------------------------------------------------------------
// Kernel 2: flash attention over i.
// Block = 8 waves; wave w owns j-tile [j0, j0+16) and the full 256-c accumulator
// (16 f32 16x16 C tiles). Stream i in chunks of 32; q (2KB) and v (16KB) tiles
// double-buffered in LDS via async loads, prefetched one step ahead.
// ---------------------------------------------------------------------------
__global__ __launch_bounds__(256)
void attn_kernel(const float* __restrict__ x, const float* __restrict__ gamma,
                 const _Float16* __restrict__ qh, const _Float16* __restrict__ kh,
                 const _Float16* __restrict__ vh, float* __restrict__ out)
{
  __shared__ alignas(16) unsigned sQ2[2][32 * 16];    // 2 x 2KB
  __shared__ alignas(16) unsigned sV2[2][256 * 16];   // 2 x 16KB

  const int lane  = threadIdx.x & 31;
  const int wv    = threadIdx.x >> 5;
  const int b     = blockIdx.y;
  const int j0    = blockIdx.x * 128 + wv * 16;
  const int lhalf = lane & 15;
  const bool lo   = lane < 16;
  const int jcol  = j0 + lhalf;

  const _Float16* qb = qh + (size_t)b * NPIX * CQN;
  const _Float16* kb = kh + (size_t)b * NPIX * CQN;
  const _Float16* vb = vh + (size_t)b * CDIM * NPIX;

  // B-matrix fragment of k for this wave's j tile: constant over the whole i loop.
  Frag16 Bk;
  {
    const unsigned* krow = (const unsigned*)(kb + (size_t)jcol * CQN);
#pragma unroll
    for (int n = 0; n < 8; ++n) Bk.u[n] = krow[lo ? (n < 4 ? n : n + 4) : (n < 4 ? n + 4 : n + 8)];
  }

  float m_run = -__builtin_inff();
  float l_run = 0.f;
  v8f acc[16];
#pragma unroll
  for (int t = 0; t < 16; ++t) { v8f z = {}; acc[t] = z; }

  const uint4* qg4 = (const uint4*)(const void*)qb;
  const uint4* vg4 = (const uint4*)(const void*)vb;

  // prologue: start async fetch of tile 0
  issue_tile_loads(qg4, vg4, 0, sQ2[0], sV2[0], threadIdx.x);

  for (int it = 0; it < NPIX / 32; ++it) {
    const int cur = it & 1;
    asm volatile("s_wait_asynccnt 0" ::: "memory");   // own async loads done
    __syncthreads();                                  // everyone's loads done; prev
                                                      // compute on other buffer done
    if (it + 1 < NPIX / 32)                           // prefetch next tile (overlaps
      issue_tile_loads(qg4, vg4, (it + 1) * 32,       //  with compute below)
                       sQ2[cur ^ 1], sV2[cur ^ 1], threadIdx.x);

    const unsigned* sQ = sQ2[cur];
    const unsigned* sV = sV2[cur];

    // QK: A = q^T (two 16x32 i-subtiles), B = k tile. s_ij already in log2 domain.
    Frag16 A0, A1;
#pragma unroll
    for (int n = 0; n < 8; ++n) {
      int dw = lo ? (n < 4 ? n : n + 4) : (n < 4 ? n + 4 : n + 8);
      A0.u[n] = sQ[lhalf * 16 + dw];
      A1.u[n] = sQ[(16 + lhalf) * 16 + dw];
    }
    v8f zc = {};
    v8f s0 = __builtin_amdgcn_wmma_f32_16x16x32_f16(false, A0.v, false, Bk.v,
                                                    (short)0, zc, false, false);
    v8f s1 = __builtin_amdgcn_wmma_f32_16x16x32_f16(false, A1.v, false, Bk.v,
                                                    (short)0, zc, false, false);

    // Block softmax update. C/D layout: lane L and L+16 share column j = L&15,
    // so one shfl_xor(16) completes the per-column reduction.
    float bm = -__builtin_inff();
#pragma unroll
    for (int r = 0; r < 8; ++r) { bm = fmaxf(bm, s0[r]); bm = fmaxf(bm, s1[r]); }
    bm = fmaxf(bm, __shfl_xor(bm, 16, 32));
    float m_new = fmaxf(m_run, bm);

    // Rescale only when the running max actually moved (rare after warmup).
    // Wave-uniform branch: s_cbranch, EXEC untouched (WMMA needs EXEC all-1s).
    if (__any(bm > m_run)) {
      float scale = exp2f(m_run - m_new);
      l_run *= scale;
#pragma unroll
      for (int t = 0; t < 16; ++t)
#pragma unroll
        for (int r = 0; r < 8; ++r) acc[t][r] *= scale;
    }
    m_run = m_new;

    float p0[8], p1[8], psum = 0.f;
#pragma unroll
    for (int r = 0; r < 8; ++r) {
      p0[r] = exp2f(s0[r] - m_new);
      p1[r] = exp2f(s1[r] - m_new);
      psum += p0[r] + p1[r];
    }
    psum += __shfl_xor(psum, 16, 32);
    l_run += psum;

    // Repack p (C/D f32 layout) into the 32x16 f16 B layout.
    Frag16 Bp;
#pragma unroll
    for (int n = 0; n < 8; ++n) {
      int r0 = (n < 4) ? 2 * n : 2 * n - 8;
      float a0 = p0[r0], a1 = p0[r0 + 1];
      float c0 = p1[r0], c1 = p1[r0 + 1];
      float xa0 = __shfl_xor(a0, 16, 32), xa1 = __shfl_xor(a1, 16, 32);
      float xc0 = __shfl_xor(c0, 16, 32), xc1 = __shfl_xor(c1, 16, 32);
      float v0, v1;
      if (n < 4) { v0 = lo ? a0 : xc0; v1 = lo ? a1 : xc1; }
      else       { v0 = lo ? xa0 : c0; v1 = lo ? xa1 : c1; }
      Bp.h[2*n]   = (_Float16)v0;
      Bp.h[2*n+1] = (_Float16)v1;
    }

    // PV: 16 c-tiles, A = v (16x32), B = p, K = 32 i values.
#pragma unroll
    for (int ct = 0; ct < 16; ++ct) {
      Frag16 Av;
#pragma unroll
      for (int n = 0; n < 8; ++n) {
        int dw = lo ? n : n + 8;
        Av.u[n] = sV[(ct * 16 + lhalf) * 16 + dw];
      }
      acc[ct] = __builtin_amdgcn_wmma_f32_16x16x32_f16(false, Av.v, false, Bp.v,
                                                       (short)0, acc[ct], false, false);
    }
  }

  // Finalize: o = gamma * acc / l + x
  const float g   = gamma[0];
  const float inv = 1.f / l_run;
  const float* xb2 = x   + (size_t)b * CDIM * NPIX;
  float*       ob  = out + (size_t)b * CDIM * NPIX;
#pragma unroll
  for (int ct = 0; ct < 16; ++ct)
#pragma unroll
    for (int r = 0; r < 8; ++r) {
      int c = ct * 16 + r + (lo ? 0 : 8);
      size_t idx = (size_t)c * NPIX + jcol;
      ob[idx] = g * acc[ct][r] * inv + xb2[idx];
    }
}

// ---------------------------------------------------------------------------
extern "C" void kernel_launch(void* const* d_in, const int* in_sizes, int n_in,
                              void* d_out, int out_size, void* d_ws, size_t ws_size,
                              hipStream_t stream) {
  const float* x     = (const float*)d_in[0];
  const float* w     = (const float*)d_in[1];
  const float* gamma = (const float*)d_in[2];
  float* out = (float*)d_out;

  // Workspace: q/k f16 [b][n][32] (1.125 MB each), v f16 [b][c][n] (9 MB). ~11.3 MB total.
  char* ws = (char*)d_ws;
  const size_t qk_bytes = (size_t)2 * NPIX * CQN * sizeof(_Float16);
  _Float16* qh = (_Float16*)ws;
  _Float16* kh = (_Float16*)(ws + qk_bytes);
  _Float16* vh = (_Float16*)(ws + 2 * qk_bytes);

  // 2 * 20 * 576 wave-tiles / 8 waves per block
  qkv_proj_kernel<<<2880, 256, 0, stream>>>(x, w, qh, kh, vh);
  // 72 j-blocks (128 cols each) x 2 batches
  attn_kernel<<<dim3(72, 2), 256, 0, stream>>>(x, gamma, qh, kh, vh, out);
}
